// SimpleRNN_86234353369124
// MI455X (gfx1250) — matrix operand order repaired and checked by reading, not compiled, not run on hardware
//
#include <hip/hip_runtime.h>
#include <hip/hip_bf16.h>

// ---------------------------------------------------------------------------
// SimpleRNN for MI455X (gfx1250, wave32, WMMA + TDM)
//   h_t = tanh(W_hh h_{t-1} + W_xh emb[x_t] + b_xh + b_hh)
// bf16 WMMA (16x16x32), f32 accumulate.
//   K1: convert weights to bf16, zero h0 + barrier counter
//   K2: gather + convert embeddings to bf16 once
//   K3: input-projection GEMM 32768x1024 (K=512), 4 col-tiles/wave,
//       mov-free ping-pong pipelined K loop
//   K4: ONE persistent kernel for all 512 recurrence steps:
//       - 16 blocks, each owns 64 hidden columns
//       - W_hh slice (128 KB) DMA'd into LDS once via TENSOR_LOAD_TO_LDS
//       - global atomic spin-barrier between timesteps (no kernel relaunch)
// ---------------------------------------------------------------------------

typedef __attribute__((ext_vector_type(16))) __bf16 v16bf;
typedef __attribute__((ext_vector_type(8)))  float  v8f;
typedef __attribute__((ext_vector_type(4)))  unsigned v4u;
typedef __attribute__((ext_vector_type(8)))  int      v8i;
typedef __attribute__((ext_vector_type(4)))  int      v4i;

#define N_BATCH 64
#define SEQ_L   512
#define EMBED   512
#define HIDDEN  1024
#define RBLOCKS 16   // blocks in persistent recurrence kernel

union Frag {
  v16bf v;
  uint4 q[2];
};

union Pack8 {
  __bf16 b[8];
  uint4  q;
};

__device__ __forceinline__ __bf16 f2bf(float f) { return (__bf16)f; }

// Fragment loads honoring WMMA 16-bit layouts.
// A (16x32): lane m<16 holds row m K={0..7,16..23}; lane m+16 K={8..15,24..31}.
// B (32x16): lane n<16 holds col n K={0..7,16..23}; lane n+16 K={8..15,24..31}.
#define LOAD_A(Adst, arow, kb)                                              \
  do {                                                                      \
    (Adst).q[0] = *(const uint4*)((arow) + (kb) +      (hi ? 8 : 0));       \
    (Adst).q[1] = *(const uint4*)((arow) + (kb) + 16 + (hi ? 8 : 0));       \
  } while (0)

#define LOAD_B(Bdst, brow, kb)                                              \
  do {                                                                      \
    (Bdst).q[0] = *(const uint4*)((brow) + (kb) +      (hi ? 8 : 0));       \
    (Bdst).q[1] = *(const uint4*)((brow) + (kb) + 16 + (hi ? 8 : 0));       \
  } while (0)

#define WMMA_BF16(acc, A, B)                                                \
  (acc) = __builtin_amdgcn_wmma_f32_16x16x32_bf16(                          \
      false, (A).v, false, (B).v, (short)0, (acc), false, false)

// ---------------------------------------------------------------------------
// K1: weight conversion f32 -> bf16, zero h0 and the barrier counter.
// ---------------------------------------------------------------------------
__global__ void __launch_bounds__(256)
rnn_prep(const float* __restrict__ Wxh, const float* __restrict__ Whh,
         __bf16* __restrict__ WxhB, __bf16* __restrict__ WhhB,
         __bf16* __restrict__ h0, unsigned* __restrict__ bar) {
  const int idx    = blockIdx.x * blockDim.x + threadIdx.x;
  const int stride = gridDim.x * blockDim.x;
  if (idx == 0) *bar = 0u;
  for (int i = idx; i < HIDDEN * EMBED;  i += stride) WxhB[i] = f2bf(Wxh[i]);
  for (int i = idx; i < HIDDEN * HIDDEN; i += stride) WhhB[i] = f2bf(Whh[i]);
  for (int i = idx; i < N_BATCH * HIDDEN; i += stride) h0[i] = f2bf(0.0f);
}

// ---------------------------------------------------------------------------
// K2: gather + convert embeddings once.  Xe[r,e] = bf16(emb[X[n,l],e]), r=l*64+n
// ---------------------------------------------------------------------------
__global__ void __launch_bounds__(256)
rnn_gather(const int* __restrict__ X, const float* __restrict__ emb,
           __bf16* __restrict__ Xe) {
  const int tid = blockIdx.x * 256 + threadIdx.x;   // 2,097,152 chunks of 8
  const int r   = tid >> 6;
  const int c8  = (tid & 63) << 3;
  const int l = r >> 6;
  const int n = r & 63;
  const int token = X[n * SEQ_L + l];
  const float* src = emb + (size_t)token * EMBED + c8;
  float4 f0 = *(const float4*)(src);
  float4 f1 = *(const float4*)(src + 4);
  Pack8 p;
  p.b[0] = f2bf(f0.x); p.b[1] = f2bf(f0.y); p.b[2] = f2bf(f0.z); p.b[3] = f2bf(f0.w);
  p.b[4] = f2bf(f1.x); p.b[5] = f2bf(f1.y); p.b[6] = f2bf(f1.z); p.b[7] = f2bf(f1.w);
  *(uint4*)(Xe + (size_t)r * EMBED + c8) = p.q;
}

// ---------------------------------------------------------------------------
// K3: input projection GEMM, 4 col-tiles per wave.
// Mov-free ping-pong pipeline: two fragment sets, no rotation copies.
// ---------------------------------------------------------------------------
struct PSet { Frag A, B0, B1, B2, B3; };

#define P_LOAD(S, kb)                                                       \
  do {                                                                      \
    LOAD_A((S).A, arow, (kb));                                              \
    LOAD_B((S).B0, brow0, (kb)); LOAD_B((S).B1, brow1, (kb));               \
    LOAD_B((S).B2, brow2, (kb)); LOAD_B((S).B3, brow3, (kb));               \
  } while (0)

#define P_WMMA(S)                                                           \
  do {                                                                      \
    WMMA_BF16(acc0, (S).A, (S).B0); WMMA_BF16(acc1, (S).A, (S).B1);         \
    WMMA_BF16(acc2, (S).A, (S).B2); WMMA_BF16(acc3, (S).A, (S).B3);         \
  } while (0)

__global__ void __launch_bounds__(256)
rnn_proj(const __bf16* __restrict__ Xe, const __bf16* __restrict__ WxhB,
         const float* __restrict__ bxh, const float* __restrict__ bhh,
         float* __restrict__ Xp) {
  const int lane  = threadIdx.x & 31;
  const int wave  = threadIdx.x >> 5;
  const int wtile = blockIdx.x * 8 + wave;   // 32768 waves
  const int ctg   = wtile & 15;
  const int rt    = wtile >> 4;
  const int hi    = lane >> 4;
  const int ln    = lane & 15;

  const __bf16* arow = Xe + (size_t)(rt * 16 + ln) * EMBED;
  const int bcol0 = ctg * 64 + ln;
  const __bf16* brow0 = WxhB + (size_t)(bcol0)      * EMBED;
  const __bf16* brow1 = WxhB + (size_t)(bcol0 + 16) * EMBED;
  const __bf16* brow2 = WxhB + (size_t)(bcol0 + 32) * EMBED;
  const __bf16* brow3 = WxhB + (size_t)(bcol0 + 48) * EMBED;

  v8f acc0 = {}, acc1 = {}, acc2 = {}, acc3 = {};
  PSet S0, S1;
  P_LOAD(S0, 0);
  P_LOAD(S1, 32);
  #pragma unroll 1
  for (int kb = 64; kb < EMBED; kb += 64) {
    P_WMMA(S0); P_LOAD(S0, kb);        // consume set0, refill for kb
    P_WMMA(S1); P_LOAD(S1, kb + 32);   // consume set1, refill for kb+32
  }
  P_WMMA(S0);                          // last two K-steps
  P_WMMA(S1);

  float* crow = Xp + ((size_t)(rt * 16 + hi * 8)) * HIDDEN;
  v8f* accs[4] = {&acc0, &acc1, &acc2, &acc3};
  #pragma unroll
  for (int i = 0; i < 4; ++i) {
    const int col = bcol0 + 16 * i;
    const float bias = bxh[col] + bhh[col];
    #pragma unroll
    for (int v = 0; v < 8; ++v)
      crow[(size_t)v * HIDDEN + col] = (*accs[i])[v] + bias;
  }
}

// ---------------------------------------------------------------------------
// K4: persistent recurrence kernel — all 512 timesteps in one launch.
// 16 blocks x 256 threads. Block b owns hidden columns [b*64, b*64+64).
// Its 64x1024 bf16 slice of Whh^T (128 KB) is staged into LDS once via the
// Tensor Data Mover, then every step reads B-fragments from LDS.
// Grid-wide sync between steps via monotonic atomic counter in ws.
// ---------------------------------------------------------------------------
struct RSet { Frag A, B0, B1; };

#define R_LOAD(S, kb)                                                       \
  do {                                                                      \
    LOAD_A((S).A, arow, (kb));                                              \
    LOAD_B((S).B0, brow0, (kb)); LOAD_B((S).B1, brow1, (kb));               \
  } while (0)

#define R_WMMA(S)                                                           \
  do {                                                                      \
    WMMA_BF16(acc0, (S).A, (S).B0); WMMA_BF16(acc1, (S).A, (S).B1);         \
  } while (0)

__global__ void __launch_bounds__(256)
rnn_recur(const __bf16* __restrict__ WhhB, const float* __restrict__ Xp,
          __bf16* __restrict__ h0, __bf16* __restrict__ h1,
          float* __restrict__ out, unsigned* __restrict__ bar) {
  extern __shared__ __bf16 WhhL[];           // 64 * 1024 bf16 = 128 KB
  const int lane = threadIdx.x & 31;
  const int wave = threadIdx.x >> 5;
  const int hi   = lane >> 4;
  const int ln   = lane & 15;
  const int colbase = blockIdx.x * 64;

  // ---- TDM: DMA this block's 64-row x 1024-col bf16 Whh^T slice into LDS ----
  // 2D tensor: dims 1024x1024 bf16, tile = 64 rows x 1024 elems at row colbase.
  if (threadIdx.x < 32) {
    const unsigned long long gaddr =
        (unsigned long long)(const void*)(WhhB + (size_t)colbase * HIDDEN);
    // D# group 0: count=1 | lds_addr=0 (start of our LDS alloc) | global addr | type=2
    v4u g0;
    g0.x = 1u;                                           // count=1 (valid)
    g0.y = 0u;                                           // lds_addr
    g0.z = (unsigned)(gaddr & 0xFFFFFFFFull);            // addr[31:0]
    g0.w = ((unsigned)(gaddr >> 32) & 0x1FFFFFFu)        // addr[56:32]
           | 0x80000000u;                                // type=2 ("image")
    // D# group 1
    v8i g1;
    g1[0] = (int)(1u << 16);                 // data_size=1 (2 bytes)
    g1[1] = (int)((HIDDEN & 0xFFFFu) << 16); // tensor_dim0[15:0]=1024
    g1[2] = (int)((HIDDEN & 0xFFFFu) << 16); // tensor_dim0 hi=0 | tensor_dim1[15:0]=1024
    g1[3] = (int)((HIDDEN & 0xFFFFu) << 16); // tensor_dim1 hi=0 | tile_dim0=1024
    g1[4] = 64;                              // tile_dim1=64 rows (tile_dim2=0)
    g1[5] = HIDDEN;                          // tensor_dim0_stride=1024
    g1[6] = 0;                               // stride0 hi | tensor_dim1_stride lo
    g1[7] = 0;
    v4i gz = {0, 0, 0, 0};                   // groups 2/3: unused (2D tensor)
#if __clang_major__ >= 23
    __builtin_amdgcn_tensor_load_to_lds(g0, g1, gz, gz, (v8i){0,0,0,0,0,0,0,0}, 0);
#else
    __builtin_amdgcn_tensor_load_to_lds(g0, g1, gz, gz, 0);
#endif
    __builtin_amdgcn_s_wait_tensorcnt(0);
  }
  __syncthreads();

  const int rt = wave & 3;                    // batch tile
  const int c0 = (wave >> 2) * 32 + ln;       // local B column (lane's)
  const __bf16* brow0 = WhhL + (size_t)(c0)      * HIDDEN;
  const __bf16* brow1 = WhhL + (size_t)(c0 + 16) * HIDDEN;
  const int gcol0 = colbase + (wave >> 2) * 32 + ln;

  __bf16* bufs[2] = { h0, h1 };

  for (int t = 0; t < SEQ_L; ++t) {
    const __bf16* hprev = bufs[t & 1];
    __bf16*       hnext = bufs[(t + 1) & 1];
    const float*  Xpt   = Xp + (size_t)t * N_BATCH * HIDDEN;
    const __bf16* arow  = hprev + (size_t)(rt * 16 + ln) * HIDDEN;

    v8f acc0 = {}, acc1 = {};
    RSet S0, S1;
    R_LOAD(S0, 0);
    R_LOAD(S1, 32);
    #pragma unroll 1
    for (int kb = 64; kb < HIDDEN; kb += 64) {
      R_WMMA(S0); R_LOAD(S0, kb);
      R_WMMA(S1); R_LOAD(S1, kb + 32);
    }
    R_WMMA(S0);
    R_WMMA(S1);

    // Epilogue: tanh(acc + Xp_t), write bf16 h_next (+ f32 out at last step)
    const int row0 = rt * 16 + hi * 8;
    v8f* accs[2] = {&acc0, &acc1};
    #pragma unroll
    for (int i = 0; i < 2; ++i) {
      const int col = gcol0 + 16 * i;
      #pragma unroll
      for (int v = 0; v < 8; ++v) {
        const size_t o = (size_t)(row0 + v) * HIDDEN + col;
        float val = tanhf((*accs[i])[v] + Xpt[o]);
        hnext[o] = f2bf(val);
        if (t == SEQ_L - 1) out[o] = val;
      }
    }

    // ---- grid-wide barrier (monotonic counter: 16 arrivals per step) ----
    __threadfence();                          // make h_next visible device-wide
    __syncthreads();                          // whole block done with step t
    if (threadIdx.x == 0) {
      __hip_atomic_fetch_add(bar, 1u, __ATOMIC_ACQ_REL, __HIP_MEMORY_SCOPE_AGENT);
      while (__hip_atomic_load(bar, __ATOMIC_ACQUIRE, __HIP_MEMORY_SCOPE_AGENT)
             < (unsigned)(RBLOCKS * (t + 1))) {
        __builtin_amdgcn_s_sleep(2);
      }
    }
    __syncthreads();                          // release whole block into step t+1
  }
}

// ---------------------------------------------------------------------------
extern "C" void kernel_launch(void* const* d_in, const int* in_sizes, int n_in,
                              void* d_out, int out_size, void* d_ws, size_t ws_size,
                              hipStream_t stream) {
  (void)in_sizes; (void)n_in; (void)out_size; (void)ws_size;

  const int*   X   = (const int*)  d_in[0];
  const float* emb = (const float*)d_in[1];
  const float* Wxh = (const float*)d_in[2];
  const float* bxh = (const float*)d_in[3];
  const float* Whh = (const float*)d_in[4];
  const float* bhh = (const float*)d_in[5];
  float* out = (float*)d_out;

  // Workspace layout (all 256B-aligned)
  char* ws = (char*)d_ws;
  __bf16* WxhB  = (__bf16*)ws;  ws += (size_t)HIDDEN * EMBED  * 2;        // 1 MB
  __bf16* WhhB  = (__bf16*)ws;  ws += (size_t)HIDDEN * HIDDEN * 2;        // 2 MB
  __bf16* hbuf0 = (__bf16*)ws;  ws += (size_t)N_BATCH * HIDDEN * 2;       // 128 KB
  __bf16* hbuf1 = (__bf16*)ws;  ws += (size_t)N_BATCH * HIDDEN * 2;       // 128 KB
  unsigned* bar = (unsigned*)ws; ws += 256;                               // barrier
  __bf16* Xe    = (__bf16*)ws;  ws += (size_t)N_BATCH * SEQ_L * EMBED * 2;// 32 MB
  float*  Xp    = (float*)ws;   // 128 MB

  rnn_prep<<<256, 256, 0, stream>>>(Wxh, Whh, WxhB, WhhB, hbuf0, bar);
  rnn_gather<<<8192, 256, 0, stream>>>(X, emb, Xe);
  rnn_proj<<<4096, 256, 0, stream>>>(Xe, WxhB, bxh, bhh, Xp);

  // Persistent recurrence: 16 blocks, 128 KB dynamic LDS each.
  rnn_recur<<<RBLOCKS, 256, 64 * HIDDEN * sizeof(__bf16), stream>>>(
      WhhB, Xp, hbuf0, hbuf1, out, bar);
}
